// Net_15083925144150
// MI455X (gfx1250) — matrix operand order repaired and checked by reading, not compiled, not run on hardware
//
#include <hip/hip_runtime.h>
#include <type_traits>

typedef __attribute__((ext_vector_type(16))) __bf16 v16bf;
typedef __attribute__((ext_vector_type(8)))  __bf16 bf16x8;
typedef __attribute__((ext_vector_type(8)))  float  v8f;
typedef __attribute__((ext_vector_type(4)))  unsigned int u32x4;
typedef __attribute__((ext_vector_type(8)))  int i32x8;
typedef __attribute__((ext_vector_type(4)))  int i32x4;

union BFrag { v16bf v; bf16x8 h[2]; };

#define TILE_M 128
#define TILE_N 128
#define TILE_K 32
#define LDS_STRIDE 40   // 80B rows: 16B-aligned, odd multiple of 16B to spread banks

#if defined(__HIP_DEVICE_COMPILE__) && \
    __has_builtin(__builtin_amdgcn_tensor_load_to_lds) && \
    __has_builtin(__builtin_amdgcn_s_wait_tensorcnt)
#define USE_TDM 1
#else
#define USE_TDM 0
#endif

#if USE_TDM
// ---------------------------------------------------------------------------
// TDM: DMA a 128-row x 32-col bf16 tile (row stride lda elems) into LDS with
// 16B padding after every 64B row -> LDS row stride 80B (= LDS_STRIDE bf16).
// D# layout per CDNA5 ISA 8.3/8.4 (2D tile, groups 2/3 zero).
// ---------------------------------------------------------------------------
__device__ __forceinline__ void tdm_load_tile(const __bf16* gsrc, const void* lds_dst,
                                              int lda)
{
    const unsigned long long ga = (unsigned long long)(size_t)gsrc;
    u32x4 g0;
    g0[0] = 1u;                                                  // count=1, user D#
    g0[1] = (unsigned)(size_t)lds_dst;                           // lds_addr (LDS byte addr)
    g0[2] = (unsigned)(ga & 0xFFFFFFFFull);                      // global_addr[31:0]
    g0[3] = (unsigned)((ga >> 32) & 0x01FFFFFFull) | (2u << 30); // addr[56:32] | type=2
    i32x8 g1;
    g1[0] = (1 << 16)     // data_size = 2 bytes
          | (1 << 20)     // pad_enable
          | (3 << 22)     // pad_interval: 16 DWORDs (one 64B row) between pads
          | (3 << 25);    // pad_amount: 4 DWORDs (16B) -> 80B LDS row stride
    g1[1] = TILE_K << 16;                // tensor_dim0 = 32 (bits 79:48, low half)
    g1[2] = TILE_M << 16;                // tensor_dim1 = 128 (bits 111:80, low half)
    g1[3] = TILE_K << 16;                // tile_dim0 = 32 (bits 127:112)
    g1[4] = TILE_M;                      // tile_dim1 = 128; tile_dim2 = 0
    g1[5] = lda;                         // tensor_dim0_stride[31:0] (elements)
    g1[6] = 0;                           // stride hi16 | dim1_stride lo16
    g1[7] = 0;
    i32x4 z4 = {0, 0, 0, 0};
#if __has_include(<hip/amd_detail/amd_gfx1250_TDM.h>)
    i32x8 z8 = {0, 0, 0, 0, 0, 0, 0, 0};
    __builtin_amdgcn_tensor_load_to_lds(g0, g1, z4, z4, z8, 0);  // 6-arg toolchain
#else
    __builtin_amdgcn_tensor_load_to_lds(g0, g1, z4, z4, 0);      // ROCm 7.2: 5-arg
#endif
}
#endif

// ---------------------------------------------------------------------------
// Weight transpose+convert: Wt[N][Kpad] (bf16, zero-padded) from W[Kreal][N] f32
// ---------------------------------------------------------------------------
__global__ __launch_bounds__(128)
void transpose_w(const float* __restrict__ W, __bf16* __restrict__ Wt,
                 int N, int Kreal, int Kpad)
{
    const int n  = blockIdx.y * 128 + threadIdx.x;
    const int k0 = blockIdx.x * 32;
    __bf16* o = Wt + (size_t)n * Kpad + k0;
    #pragma unroll
    for (int j = 0; j < 32; ++j) {
        const int k = k0 + j;
        float v = (k < Kreal) ? W[(size_t)k * N + n] : 0.f;   // coalesced across lanes
        o[j] = (__bf16)v;
    }
}

// ---------------------------------------------------------------------------
// f32->bf16 staging loader (global -> registers)
// ---------------------------------------------------------------------------
__device__ __forceinline__ void stage_load_Af32(const float* __restrict__ A, int lda,
                                                int Kreal, size_t gr, int k0, int sk,
                                                bf16x8& a0, bf16x8& a1)
{
    if (k0 + 31 < Kreal) {   // block-uniform: scalar branch, no divergence
        const float4* s4 = reinterpret_cast<const float4*>(A + gr * (size_t)lda + k0 + sk);
        float4 f0 = s4[0], f1 = s4[1], f2 = s4[2], f3 = s4[3];
        a0[0]=(__bf16)f0.x; a0[1]=(__bf16)f0.y; a0[2]=(__bf16)f0.z; a0[3]=(__bf16)f0.w;
        a0[4]=(__bf16)f1.x; a0[5]=(__bf16)f1.y; a0[6]=(__bf16)f1.z; a0[7]=(__bf16)f1.w;
        a1[0]=(__bf16)f2.x; a1[1]=(__bf16)f2.y; a1[2]=(__bf16)f2.z; a1[3]=(__bf16)f2.w;
        a1[4]=(__bf16)f3.x; a1[5]=(__bf16)f3.y; a1[6]=(__bf16)f3.z; a1[7]=(__bf16)f3.w;
    } else {                 // only the final K-tile of f32 inputs
        #pragma unroll
        for (int j = 0; j < 8; ++j) {
            const int k = k0 + sk + j;
            a0[j] = (__bf16)((k < Kreal) ? A[gr * (size_t)lda + k] : 0.f);
        }
        #pragma unroll
        for (int j = 0; j < 8; ++j) {
            const int k = k0 + sk + 8 + j;
            a1[j] = (__bf16)((k < Kreal) ? A[gr * (size_t)lda + k] : 0.f);
        }
    }
}

__device__ __forceinline__ void stage_load_W(const __bf16* __restrict__ Wt, int ldw,
                                             int n0, int sr, int k0, int sk,
                                             bf16x8& w0, bf16x8& w1)
{
    const bf16x8* s = reinterpret_cast<const bf16x8*>(Wt + (size_t)(n0 + sr) * ldw + k0 + sk);
    w0 = s[0]; w1 = s[1];
}

// ---------------------------------------------------------------------------
// Encoder GEMM (f32 A, converted on the fly): C = relu(A@W + b), bf16 WMMA.
// REQUIRES rows % 128 == 0. Software pipeline via registers.
// ---------------------------------------------------------------------------
__global__ __launch_bounds__(256)
void gemm_bias_relu_f32(const float* __restrict__ A, int lda,
                        const __bf16* __restrict__ Wt, int ldw,
                        const float* __restrict__ bias,
                        float* __restrict__ C, int ldc,
                        int Kreal, int Ktiles)
{
    __shared__ __bf16 As[TILE_M][LDS_STRIDE];
    __shared__ __bf16 Ws[TILE_N][LDS_STRIDE];   // transposed: Ws[n][k]

    const int m0   = blockIdx.x * TILE_M;
    const int n0   = blockIdx.y * TILE_N;
    const int lane = threadIdx.x & 31;
    const int wid  = threadIdx.x >> 5;
    const int sr   = threadIdx.x >> 1;         // 0..127 staging row / n-index
    const int sk   = (threadIdx.x & 1) * 16;   // 0 or 16
    const size_t gr = (size_t)m0 + sr;

    v8f acc[8] = {};
    bf16x8 a0, a1, w0, w1;

    stage_load_Af32(A, lda, Kreal, gr, 0, sk, a0, a1);
    stage_load_W(Wt, ldw, n0, sr, 0, sk, w0, w1);
    *reinterpret_cast<bf16x8*>(&As[sr][sk])     = a0;
    *reinterpret_cast<bf16x8*>(&As[sr][sk + 8]) = a1;
    *reinterpret_cast<bf16x8*>(&Ws[sr][sk])     = w0;
    *reinterpret_cast<bf16x8*>(&Ws[sr][sk + 8]) = w1;
    __syncthreads();

    const int arow = (wid << 4) + (lane & 15);
    const int kseg = lane >> 4;
    const int bcol = lane & 15;

    for (int kt = 0; kt < Ktiles; ++kt) {
        const bool more = (kt + 1) < Ktiles;
        if (more) {
            const int k0 = (kt + 1) * TILE_K;
            stage_load_Af32(A, lda, Kreal, gr, k0, sk, a0, a1);
            stage_load_W(Wt, ldw, n0, sr, k0, sk, w0, w1);
        }
        if (kt + 2 < Ktiles)
            __builtin_prefetch(A + gr * (size_t)lda + (size_t)(kt + 2) * TILE_K, 0, 1);

        BFrag a;
        a.h[0] = *reinterpret_cast<const bf16x8*>(&As[arow][kseg * 8]);
        a.h[1] = *reinterpret_cast<const bf16x8*>(&As[arow][16 + kseg * 8]);
        BFrag bfr[8];
        #pragma unroll
        for (int nt = 0; nt < 8; ++nt) {
            const __bf16* wp = &Ws[nt * 16 + bcol][kseg * 8];
            bfr[nt].h[0] = *reinterpret_cast<const bf16x8*>(wp);
            bfr[nt].h[1] = *reinterpret_cast<const bf16x8*>(wp + 16);
        }
        #pragma unroll
        for (int nt = 0; nt < 8; ++nt)
            acc[nt] = __builtin_amdgcn_wmma_f32_16x16x32_bf16(
                false, a.v, false, bfr[nt].v, (short)0, acc[nt], false, false);

        __syncthreads();
        if (more) {
            *reinterpret_cast<bf16x8*>(&As[sr][sk])     = a0;
            *reinterpret_cast<bf16x8*>(&As[sr][sk + 8]) = a1;
            *reinterpret_cast<bf16x8*>(&Ws[sr][sk])     = w0;
            *reinterpret_cast<bf16x8*>(&Ws[sr][sk + 8]) = w1;
            __syncthreads();
        }
    }

    #pragma unroll
    for (int nt = 0; nt < 8; ++nt) {
        const int col = n0 + nt * 16 + bcol;
        const float bv = bias[col];
        #pragma unroll
        for (int j = 0; j < 8; ++j) {
            const size_t row = (size_t)m0 + (wid << 4) + kseg * 8 + j;
            float v = acc[nt][j] + bv;
            C[row * (size_t)ldc + col] = v > 0.f ? v : 0.f;
        }
    }
}

// ---------------------------------------------------------------------------
// bf16-A GEMM (pre-padded A): TDM double-buffered staging when available,
// register pipeline fallback otherwise. One barrier per K-step with TDM.
// ---------------------------------------------------------------------------
__global__ __launch_bounds__(256)
void gemm_bias_relu_bf16(const __bf16* __restrict__ A, int lda,
                         const __bf16* __restrict__ Wt, int ldw,
                         const float* __restrict__ bias,
                         float* __restrict__ C, int ldc, int Ktiles)
{
    const int m0   = blockIdx.x * TILE_M;
    const int n0   = blockIdx.y * TILE_N;
    const int lane = threadIdx.x & 31;
    const int wid  = threadIdx.x >> 5;
    const int arow = (wid << 4) + (lane & 15);
    const int kseg = lane >> 4;
    const int bcol = lane & 15;

    v8f acc[8] = {};

#if USE_TDM
    __shared__ __bf16 As2[2][TILE_M][LDS_STRIDE];
    __shared__ __bf16 Ws2[2][TILE_N][LDS_STRIDE];

    if (wid == 0) {   // TDM issue: EXEC ignored, TENSORcnt is per-wave
        tdm_load_tile(A  + (size_t)m0 * lda, &As2[0][0][0], lda);
        tdm_load_tile(Wt + (size_t)n0 * ldw, &Ws2[0][0][0], ldw);
    }
    int buf = 0;
    for (int kt = 0; kt < Ktiles; ++kt) {
        if (wid == 0) __builtin_amdgcn_s_wait_tensorcnt(0);   // tile kt landed in buf
        __syncthreads();                                      // release consumers; prior
                                                              // reads of buf^1 also done
        if (wid == 0 && (kt + 1) < Ktiles) {                  // DMA next tile into buf^1
            const int k0 = (kt + 1) * TILE_K;
            tdm_load_tile(A  + (size_t)m0 * lda + k0, &As2[buf ^ 1][0][0], lda);
            tdm_load_tile(Wt + (size_t)n0 * ldw + k0, &Ws2[buf ^ 1][0][0], ldw);
        }
        BFrag a;
        a.h[0] = *reinterpret_cast<const bf16x8*>(&As2[buf][arow][kseg * 8]);
        a.h[1] = *reinterpret_cast<const bf16x8*>(&As2[buf][arow][16 + kseg * 8]);
        BFrag bfr[8];
        #pragma unroll
        for (int nt = 0; nt < 8; ++nt) {
            const __bf16* wp = &Ws2[buf][nt * 16 + bcol][kseg * 8];
            bfr[nt].h[0] = *reinterpret_cast<const bf16x8*>(wp);
            bfr[nt].h[1] = *reinterpret_cast<const bf16x8*>(wp + 16);
        }
        #pragma unroll
        for (int nt = 0; nt < 8; ++nt)
            acc[nt] = __builtin_amdgcn_wmma_f32_16x16x32_bf16(
                false, a.v, false, bfr[nt].v, (short)0, acc[nt], false, false);
        buf ^= 1;
    }
#else
    __shared__ __bf16 As[TILE_M][LDS_STRIDE];
    __shared__ __bf16 Ws[TILE_N][LDS_STRIDE];
    const int sr = threadIdx.x >> 1;
    const int sk = (threadIdx.x & 1) * 16;
    const size_t gr = (size_t)m0 + sr;
    bf16x8 a0, a1, w0, w1;
    {
        const bf16x8* s = reinterpret_cast<const bf16x8*>(A + gr * (size_t)lda + sk);
        a0 = s[0]; a1 = s[1];
    }
    stage_load_W(Wt, ldw, n0, sr, 0, sk, w0, w1);
    *reinterpret_cast<bf16x8*>(&As[sr][sk])     = a0;
    *reinterpret_cast<bf16x8*>(&As[sr][sk + 8]) = a1;
    *reinterpret_cast<bf16x8*>(&Ws[sr][sk])     = w0;
    *reinterpret_cast<bf16x8*>(&Ws[sr][sk + 8]) = w1;
    __syncthreads();
    for (int kt = 0; kt < Ktiles; ++kt) {
        const bool more = (kt + 1) < Ktiles;
        if (more) {
            const int k0 = (kt + 1) * TILE_K;
            const bf16x8* s = reinterpret_cast<const bf16x8*>(A + gr * (size_t)lda + k0 + sk);
            a0 = s[0]; a1 = s[1];
            stage_load_W(Wt, ldw, n0, sr, k0, sk, w0, w1);
        }
        BFrag a;
        a.h[0] = *reinterpret_cast<const bf16x8*>(&As[arow][kseg * 8]);
        a.h[1] = *reinterpret_cast<const bf16x8*>(&As[arow][16 + kseg * 8]);
        BFrag bfr[8];
        #pragma unroll
        for (int nt = 0; nt < 8; ++nt) {
            const __bf16* wp = &Ws[nt * 16 + bcol][kseg * 8];
            bfr[nt].h[0] = *reinterpret_cast<const bf16x8*>(wp);
            bfr[nt].h[1] = *reinterpret_cast<const bf16x8*>(wp + 16);
        }
        #pragma unroll
        for (int nt = 0; nt < 8; ++nt)
            acc[nt] = __builtin_amdgcn_wmma_f32_16x16x32_bf16(
                false, a.v, false, bfr[nt].v, (short)0, acc[nt], false, false);
        __syncthreads();
        if (more) {
            *reinterpret_cast<bf16x8*>(&As[sr][sk])     = a0;
            *reinterpret_cast<bf16x8*>(&As[sr][sk + 8]) = a1;
            *reinterpret_cast<bf16x8*>(&Ws[sr][sk])     = w0;
            *reinterpret_cast<bf16x8*>(&Ws[sr][sk + 8]) = w1;
            __syncthreads();
        }
    }
#endif

    #pragma unroll
    for (int nt = 0; nt < 8; ++nt) {
        const int col = n0 + nt * 16 + bcol;
        const float bv = bias[col];
        #pragma unroll
        for (int j = 0; j < 8; ++j) {
            const size_t row = (size_t)m0 + (wid << 4) + kseg * 8 + j;
            float v = acc[nt][j] + bv;
            C[row * (size_t)ldc + col] = v > 0.f ? v : 0.f;
        }
    }
}

// ---------------------------------------------------------------------------
// Assemble u_in[6144][1312] (bf16, zero padded): per-user reductions + concat
// ---------------------------------------------------------------------------
__global__ __launch_bounds__(128)
void assemble_u_in(const float* __restrict__ ms, const float* __restrict__ pool,
                   const float* __restrict__ lk, const float* __restrict__ item,
                   const float* __restrict__ berry, const float* __restrict__ ab,
                   const float* __restrict__ user_x, const float* __restrict__ lk_mask,
                   const float* __restrict__ berry_mask, const float* __restrict__ types,
                   const float* __restrict__ tera, __bf16* __restrict__ out)
{
    const int n = blockIdx.x;      // 0..6143 = (b,2,6)
    const int c = threadIdx.x;     // 0..127
    __bf16* o = out + (size_t)n * 1312;

    const float* msb = ms + ((size_t)n * 4) * 128 + c;
    float mmax = msb[0];
    #pragma unroll
    for (int m = 1; m < 4; ++m) mmax = fmaxf(mmax, msb[(size_t)m * 128]);

    const float* pb = pool + ((size_t)n * 8) * 128 + c;
    float pm = 0.f;
    #pragma unroll
    for (int p = 0; p < 8; ++p) pm += pb[(size_t)p * 128];
    pm *= 0.125f;

    const float msx = fmaxf(mmax, pm);
    const float itv = 0.5f * (item[((size_t)n * 2) * 128 + c] +
                              item[((size_t)n * 2 + 1) * 128 + c]);
    const float abv = 0.5f * (ab[((size_t)n * 2) * 128 + c] +
                              ab[((size_t)n * 2 + 1) * 128 + c]);
    const float brv = berry[(size_t)n * 128 + c] * berry_mask[n];

    #pragma unroll
    for (int l = 0; l < 5; ++l)
        o[89 + l * 128 + c] =
            (__bf16)(lk[((size_t)n * 5 + l) * 128 + c] * lk_mask[(size_t)n * 5 + l]);
    o[729 + c]  = (__bf16)brv;
    o[857 + c]  = (__bf16)msx;
    o[985 + c]  = (__bf16)itv;
    o[1113 + c] = (__bf16)abv;
    if (c < 89) o[c] = (__bf16)user_x[(size_t)n * 89 + c];
    if (c < 20) {
        o[1241 + c] = (__bf16)types[(size_t)n * 20 + c];
        o[1261 + c] = (__bf16)tera[(size_t)n * 20 + c];
    }
    if (c < 31) o[1281 + c] = (__bf16)0.f;   // pad to 1312
}

// ---------------------------------------------------------------------------
// battle_full[512][2091] (bf16): [battle_x(9) | side0(17+512+512) | side1(...)]
// ---------------------------------------------------------------------------
__global__ __launch_bounds__(256)
void assemble_battle(const float* __restrict__ battle_x, const float* __restrict__ side_x,
                     const float* __restrict__ u, const int* __restrict__ active_idx,
                     __bf16* __restrict__ bf)
{
    const int b = blockIdx.x;
    const int t = threadIdx.x;
    __bf16* o = bf + (size_t)b * 2091;
    if (t < 9) o[t] = (__bf16)battle_x[(size_t)b * 9 + t];
    #pragma unroll
    for (int s = 0; s < 2; ++s) {
        const size_t base = 9 + (size_t)s * 1041;
        if (t < 17) o[base + t] = (__bf16)side_x[((size_t)b * 2 + s) * 17 + t];
        const int ai = active_idx[b * 2 + s];
        const float* ub = u + ((size_t)(b * 2 + s) * 6) * 512;
        for (int c = t; c < 512; c += 256) {
            o[base + 17 + c] = (__bf16)ub[(size_t)ai * 512 + c];
            float mx = ub[c];
            #pragma unroll
            for (int i = 1; i < 6; ++i) mx = fmaxf(mx, ub[(size_t)i * 512 + c]);
            o[base + 529 + c] = (__bf16)mx;
        }
    }
}

// xm[4096][2240] (bf16): [battle_full(2091) | opt_move(128) | tera(1) | pad]
__global__ __launch_bounds__(256)
void assemble_xm(const __bf16* __restrict__ bf, const float* __restrict__ opt_moves,
                 __bf16* __restrict__ xm)
{
    const int r = blockIdx.x;            // b*8 + m*2 + t
    const int b = r >> 3, m = (r >> 1) & 3, tt = r & 1;
    __bf16* o = xm + (size_t)r * 2240;
    const __bf16* src = bf + (size_t)b * 2091;
    for (int c = threadIdx.x; c < 2091; c += 256) o[c] = src[c];
    if (threadIdx.x < 128)
        o[2091 + threadIdx.x] =
            (__bf16)opt_moves[((size_t)b * 4 + m) * 128 + threadIdx.x];
    if (threadIdx.x == 0) o[2219] = (__bf16)(float)tt;
    for (int c = 2220 + threadIdx.x; c < 2240; c += 256) o[c] = (__bf16)0.f;
}

// xs[3072][2624] (bf16): [battle_full(2091) | u[b,0,switch_idx](512) | pad]
__global__ __launch_bounds__(256)
void assemble_xs(const __bf16* __restrict__ bf, const float* __restrict__ u,
                 const int* __restrict__ switch_idx, __bf16* __restrict__ xs)
{
    const int r = blockIdx.x;            // b*6 + j
    const int b = r / 6, j = r % 6;
    __bf16* o = xs + (size_t)r * 2624;
    const __bf16* src = bf + (size_t)b * 2091;
    const int si = switch_idx[b * 6 + j];
    const float* ub = u + (((size_t)b * 2) * 6 + si) * 512;   // side 0
    for (int c = threadIdx.x; c < 2091; c += 256) o[c] = src[c];
    for (int c = threadIdx.x; c < 512; c += 256) o[2091 + c] = (__bf16)ub[c];
    for (int c = 2603 + threadIdx.x; c < 2624; c += 256) o[c] = (__bf16)0.f;
}

// ---------------------------------------------------------------------------
// Final heads: 14 masked logits per batch element (wave-per-logit dot product)
// ---------------------------------------------------------------------------
__global__ __launch_bounds__(512)
void logits_kernel(const float* __restrict__ h_m, const float* __restrict__ h_s,
                   const float* __restrict__ W_mo2, const float* __restrict__ b_mo2,
                   const float* __restrict__ W_so2, const float* __restrict__ b_so2,
                   const float* __restrict__ move_mask, const float* __restrict__ can_tera,
                   const float* __restrict__ switch_mask, float* __restrict__ out)
{
    const int b = blockIdx.x;
    const int w = threadIdx.x >> 5;      // 16 waves; 0-7 move opts, 8-13 switch
    const int lane = threadIdx.x & 31;
    float s = 0.f;
    if (w < 8) {
        const float* h = h_m + ((size_t)b * 8 + w) * 512;
        for (int i = lane; i < 512; i += 32) s += h[i] * W_mo2[i];
    } else if (w < 14) {
        const float* h = h_s + ((size_t)b * 6 + (w - 8)) * 512;
        for (int i = lane; i < 512; i += 32) s += h[i] * W_so2[i];
    }
    #pragma unroll
    for (int off = 16; off; off >>= 1) s += __shfl_xor(s, off, 32);
    if (lane == 0) {
        if (w < 8) {
            const int m = w >> 1, tt = w & 1;
            const bool valid =
                (move_mask[b * 4 + m] > 0.f) && (tt == 0 || can_tera[b] > 0.f);
            out[(size_t)b * 14 + w] = valid ? (s + b_mo2[0]) : -__builtin_inff();
        } else if (w < 14) {
            const int j = w - 8;
            out[(size_t)b * 14 + 8 + j] =
                (switch_mask[b * 6 + j] > 0.f) ? (s + b_so2[0]) : -__builtin_inff();
        }
    }
}

// ---------------------------------------------------------------------------
// Host wiring / workspace layout (region A reused across phases)
// ---------------------------------------------------------------------------
static constexpr size_t OFF_MS    = 0;
static constexpr size_t OFF_POOL  = OFF_MS    + 24576ull * 128 * 4;
static constexpr size_t OFF_LK    = OFF_POOL  + 49152ull * 128 * 4;
static constexpr size_t OFF_ITEM  = OFF_LK    + 30720ull * 128 * 4;
static constexpr size_t OFF_BERRY = OFF_ITEM  + 12288ull * 128 * 4;
static constexpr size_t OFF_AB    = OFF_BERRY +  6144ull * 128 * 4;
static constexpr size_t END_A     = OFF_AB    + 12288ull * 128 * 4;   // 69,206,016
// phase-2 overlays of region A (region A dead after assemble_u_in)
static constexpr size_t OFF_XM    = 0;
static constexpr size_t OFF_XS    = OFF_XM + 4096ull * 2240 * 2;
static constexpr size_t OFF_HM    = OFF_XS + 3072ull * 2624 * 2;
static constexpr size_t OFF_HS    = OFF_HM + 4096ull * 512 * 4;
// non-overlapping
static constexpr size_t OFF_UIN   = END_A;                            // bf16 6144x1312
static constexpr size_t OFF_U     = OFF_UIN + 6144ull * 1312 * 2;     // f32  6144x512
static constexpr size_t OFF_BF    = OFF_U   + 6144ull * 512 * 4;      // bf16 512x2091
// pre-transposed bf16 weights [N][Kpad]
static constexpr size_t OFF_WT_MS   = OFF_BF      + 512ull * 2091 * 2;
static constexpr size_t OFF_WT_ITEM = OFF_WT_MS   + 128ull * 544 * 2;
static constexpr size_t OFF_WT_AB   = OFF_WT_ITEM + 128ull * 256 * 2;
static constexpr size_t OFF_WT_USER = OFF_WT_AB   + 128ull * 256 * 2;
static constexpr size_t OFF_WT_MO1  = OFF_WT_USER + 512ull * 1312 * 2;
static constexpr size_t OFF_WT_SO1  = OFF_WT_MO1  + 512ull * 2240 * 2;

extern "C" void kernel_launch(void* const* d_in, const int* in_sizes, int n_in,
                              void* d_out, int out_size, void* d_ws, size_t ws_size,
                              hipStream_t stream) {
    (void)in_sizes; (void)n_in; (void)out_size; (void)ws_size;
    const float* battle_x    = (const float*)d_in[0];
    const float* side_x      = (const float*)d_in[1];
    const float* user_x      = (const float*)d_in[2];
    const float* moveset     = (const float*)d_in[3];
    const float* movepool    = (const float*)d_in[4];
    const float* lookup      = (const float*)d_in[5];
    const float* lk_mask     = (const float*)d_in[6];
    const float* lastberry   = (const float*)d_in[7];
    const float* berry_mask  = (const float*)d_in[8];
    const float* items       = (const float*)d_in[9];
    const float* abilities   = (const float*)d_in[10];
    const float* types_x     = (const float*)d_in[11];
    const float* tera_types  = (const float*)d_in[12];
    const float* opt_moves   = (const float*)d_in[13];
    const float* move_mask   = (const float*)d_in[14];
    const float* can_tera    = (const float*)d_in[15];
    const float* switch_mask = (const float*)d_in[16];
    const float* W_item = (const float*)d_in[17]; const float* b_item = (const float*)d_in[18];
    const float* W_ab   = (const float*)d_in[19]; const float* b_ab   = (const float*)d_in[20];
    const float* W_ms   = (const float*)d_in[21]; const float* b_ms   = (const float*)d_in[22];
    const float* W_user = (const float*)d_in[23]; const float* b_user = (const float*)d_in[24];
    const float* W_mo1  = (const float*)d_in[25]; const float* b_mo1  = (const float*)d_in[26];
    const float* W_mo2  = (const float*)d_in[27]; const float* b_mo2  = (const float*)d_in[28];
    const float* W_so1  = (const float*)d_in[29]; const float* b_so1  = (const float*)d_in[30];
    const float* W_so2  = (const float*)d_in[31]; const float* b_so2  = (const float*)d_in[32];
    const int* active_idx = (const int*)d_in[33];
    const int* switch_idx = (const int*)d_in[34];

    char* ws = (char*)d_ws;
    float*  ms_o    = (float*)(ws + OFF_MS);
    float*  pool_o  = (float*)(ws + OFF_POOL);
    float*  lk_o    = (float*)(ws + OFF_LK);
    float*  item_o  = (float*)(ws + OFF_ITEM);
    float*  berry_o = (float*)(ws + OFF_BERRY);
    float*  ab_o    = (float*)(ws + OFF_AB);
    __bf16* u_in    = (__bf16*)(ws + OFF_UIN);
    float*  u       = (float*)(ws + OFF_U);
    __bf16* bfull   = (__bf16*)(ws + OFF_BF);
    __bf16* xm      = (__bf16*)(ws + OFF_XM);
    __bf16* xs      = (__bf16*)(ws + OFF_XS);
    float*  h_m     = (float*)(ws + OFF_HM);
    float*  h_s     = (float*)(ws + OFF_HS);
    __bf16* wt_ms   = (__bf16*)(ws + OFF_WT_MS);
    __bf16* wt_item = (__bf16*)(ws + OFF_WT_ITEM);
    __bf16* wt_ab   = (__bf16*)(ws + OFF_WT_AB);
    __bf16* wt_user = (__bf16*)(ws + OFF_WT_USER);
    __bf16* wt_mo1  = (__bf16*)(ws + OFF_WT_MO1);
    __bf16* wt_so1  = (__bf16*)(ws + OFF_WT_SO1);

    const dim3 blk(256);
    auto grid = [](int rows, int N) { return dim3((rows + 127) / 128, N / 128); };

    // phase 0: transpose+convert weights to bf16 [N][Kpad] (tiny, L2-resident)
    transpose_w<<<dim3(17, 1), dim3(128), 0, stream>>>(W_ms,   wt_ms,   128, 516,  544);
    transpose_w<<<dim3(8,  1), dim3(128), 0, stream>>>(W_item, wt_item, 128, 256,  256);
    transpose_w<<<dim3(8,  1), dim3(128), 0, stream>>>(W_ab,   wt_ab,   128, 256,  256);
    transpose_w<<<dim3(41, 4), dim3(128), 0, stream>>>(W_user, wt_user, 512, 1281, 1312);
    transpose_w<<<dim3(70, 4), dim3(128), 0, stream>>>(W_mo1,  wt_mo1,  512, 2220, 2240);
    transpose_w<<<dim3(82, 4), dim3(128), 0, stream>>>(W_so1,  wt_so1,  512, 2603, 2624);

    // phase 1: encoders (f32 A, convert-on-the-fly register pipeline)
    gemm_bias_relu_f32<<<grid(24576, 128), blk, 0, stream>>>(moveset,  516, wt_ms,   544, b_ms,   ms_o,    128, 516, 17);
    gemm_bias_relu_f32<<<grid(49152, 128), blk, 0, stream>>>(movepool, 516, wt_ms,   544, b_ms,   pool_o,  128, 516, 17);
    gemm_bias_relu_f32<<<grid(30720, 128), blk, 0, stream>>>(lookup,   516, wt_ms,   544, b_ms,   lk_o,    128, 516, 17);
    gemm_bias_relu_f32<<<grid(12288, 128), blk, 0, stream>>>(items,    256, wt_item, 256, b_item, item_o,  128, 256, 8);
    gemm_bias_relu_f32<<<grid(6144,  128), blk, 0, stream>>>(lastberry,256, wt_item, 256, b_item, berry_o, 128, 256, 8);
    gemm_bias_relu_f32<<<grid(12288, 128), blk, 0, stream>>>(abilities,256, wt_ab,   256, b_ab,   ab_o,    128, 256, 8);

    // phase 2: user block (TDM-staged bf16 GEMM)
    assemble_u_in<<<dim3(6144), dim3(128), 0, stream>>>(ms_o, pool_o, lk_o, item_o, berry_o, ab_o,
                                                        user_x, lk_mask, berry_mask,
                                                        types_x, tera_types, u_in);
    gemm_bias_relu_bf16<<<grid(6144, 512), blk, 0, stream>>>(u_in, 1312, wt_user, 1312, b_user, u, 512, 41);

    // phase 3: heads (TDM-staged bf16 GEMMs)
    assemble_battle<<<dim3(512), dim3(256), 0, stream>>>(battle_x, side_x, u, active_idx, bfull);
    assemble_xm<<<dim3(4096), dim3(256), 0, stream>>>(bfull, opt_moves, xm);
    assemble_xs<<<dim3(3072), dim3(256), 0, stream>>>(bfull, u, switch_idx, xs);
    gemm_bias_relu_bf16<<<grid(4096, 512), blk, 0, stream>>>(xm, 2240, wt_mo1, 2240, b_mo1, h_m, 512, 70);
    gemm_bias_relu_bf16<<<grid(3072, 512), blk, 0, stream>>>(xs, 2624, wt_so1, 2624, b_so1, h_s, 512, 82);

    logits_kernel<<<dim3(512), dim3(512), 0, stream>>>(h_m, h_s, W_mo2, b_mo2, W_so2, b_so2,
                                                       move_mask, can_tera, switch_mask,
                                                       (float*)d_out);
}
